// VanillaRNN_53060025975309
// MI455X (gfx1250) — matrix operand order repaired and checked by reading, not compiled, not run on hardware
//
#include <hip/hip_runtime.h>

// ---------------------------------------------------------------------------
// VanillaRNN on MI455X (gfx1250).
// Identity: h0 == 0 makes the [H,H] hidden state row-constant forever, so it
// collapses to one [H] vector v. Pipeline:
//   1) split x / W_hx into hi/lo bf16 (W transposed via LDS tiles)   [once]
//   2) xw = x @ W_hx via v_wmma_f32_16x16x32_bf16, 3-term hi/lo split
//   3) v_t = tanh(xw_t + v W_hh + b_h), t=0..255: persistent kernel, grid
//      barrier per step; W_hh (16MB) is L2-resident (192MB L2) -> the 256 x
//      16MB L2 stream is the runtime floor, spread over 128 WGPs
//   4) out = softmax(2048 * (v @ W_ph + b_p))
// ---------------------------------------------------------------------------

typedef __attribute__((ext_vector_type(16))) __bf16 bf16x16;
typedef __attribute__((ext_vector_type(8)))  __bf16 bf16x8;
typedef __attribute__((ext_vector_type(8)))  float  floatx8;

#define SEQ_LEN     256
#define INPUT_DIM   4096
#define NUM_HIDDEN  2048
#define NUM_CLASSES 10
#define RNN_BLOCKS  128

__device__ __forceinline__ __bf16 f2bf_rne(float f) {
  unsigned u = __builtin_bit_cast(unsigned, f);
  u += 0x7FFFu + ((u >> 16) & 1u);              // round-to-nearest-even
  unsigned short h = (unsigned short)(u >> 16);
  return __builtin_bit_cast(__bf16, h);
}
__device__ __forceinline__ float bf2f(__bf16 b) {
  unsigned u = ((unsigned)__builtin_bit_cast(unsigned short, b)) << 16;
  return __builtin_bit_cast(float, u);
}

// ---------------------------------------------------------------------------
// Kernel 0: zero v0 and the grid-barrier counter (every launch -> the
// persistent kernel's barrier state is deterministic across graph replays).
// ---------------------------------------------------------------------------
__global__ void init_kernel(float* __restrict__ v0, unsigned* __restrict__ bar) {
  int i = blockIdx.x * 256 + threadIdx.x;
  if (i < NUM_HIDDEN) v0[i] = 0.0f;
  if (i == 0) *bar = 0u;
}

// ---------------------------------------------------------------------------
// One-time hi/lo bf16 split of x (layout preserved: A fragments read
// K-contiguous runs, which row-major x already provides).
// ---------------------------------------------------------------------------
__global__ __launch_bounds__(256) void split_x_kernel(
    const float* __restrict__ x, __bf16* __restrict__ xhi,
    __bf16* __restrict__ xlo) {
  int i = blockIdx.x * 256 + threadIdx.x;     // 256*4096 elements
  float f = x[i];
  __bf16 h = f2bf_rne(f);
  xhi[i] = h;
  xlo[i] = f2bf_rne(f - bf2f(h));
}

// ---------------------------------------------------------------------------
// One-time hi/lo split + transpose: W_hx [4096][2048] -> WT_hi/lo [2048][4096]
// so B fragments (16 K-values at fixed N) become one contiguous 32B load.
// LDS 32x33 tile: coalesced global reads, conflict-free LDS column reads,
// coalesced b16 writes.
// ---------------------------------------------------------------------------
__global__ __launch_bounds__(256) void split_wT_kernel(
    const float* __restrict__ W, __bf16* __restrict__ wThi,
    __bf16* __restrict__ wTlo) {
  __shared__ float tile[32][33];
  const int tk0 = (blockIdx.x >> 6) * 32;     // 4096/32 = 128 K-tiles
  const int tn0 = (blockIdx.x & 63) * 32;     // 2048/32 = 64  N-tiles
  const int c  = threadIdx.x & 31;
  const int r4 = threadIdx.x >> 5;            // 0..7
#pragma unroll
  for (int j = 0; j < 4; ++j) {
    int r = r4 * 4 + j;
    tile[r][c] = W[(size_t)(tk0 + r) * NUM_HIDDEN + tn0 + c];
  }
  __syncthreads();
#pragma unroll
  for (int j = 0; j < 4; ++j) {
    int n = r4 * 4 + j;
    float f = tile[c][n];                     // = W[tk0+c][tn0+n]
    __bf16 h = f2bf_rne(f);
    size_t o = (size_t)(tn0 + n) * INPUT_DIM + tk0 + c;
    wThi[o] = h;
    wTlo[o] = f2bf_rne(f - bf2f(h));
  }
}

// ---------------------------------------------------------------------------
// GEMM v2: xw = x @ W_hx from pre-split bf16 operands. One wave per 16x16
// tile; 2048 tiles = 256 blocks x 8 waves. Per 16x16x32 K-step: 4 b128 A
// loads + 2 32B B loads + 3 WMMAs (hi*hi + hi*lo + lo*hi ~ fp32 accuracy).
// Fragment packing (contiguous K order, same in A and B):
//   A: lane = half*16+m; elem i -> K = k0+half*16+i
//   B: lane = half*16+n; elem i -> K = k0+half*16+i   (contiguous in WT)
//   C: lane = half*16+n; vgpr j -> M = half*8+j
// ---------------------------------------------------------------------------
__global__ __launch_bounds__(256) void gemm_xw_bf16_kernel(
    const __bf16* __restrict__ xhi, const __bf16* __restrict__ xlo,
    const __bf16* __restrict__ wThi, const __bf16* __restrict__ wTlo,
    float* __restrict__ xw) {
  const int lane  = threadIdx.x & 31;
  const int wid   = (blockIdx.x * 256 + threadIdx.x) >> 5;  // 0..2047
  const int tileM = wid >> 7;
  const int tileN = wid & 127;
  const int half  = lane >> 4;
  const int mn    = lane & 15;

  const __bf16* ah = xhi + (size_t)(tileM * 16 + mn) * INPUT_DIM + half * 16;
  const __bf16* al = xlo + (size_t)(tileM * 16 + mn) * INPUT_DIM + half * 16;
  const __bf16* bh = wThi + (size_t)(tileN * 16 + mn) * INPUT_DIM + half * 16;
  const __bf16* bl = wTlo + (size_t)(tileN * 16 + mn) * INPUT_DIM + half * 16;

  floatx8 acc = {};
#pragma unroll 4
  for (int k0 = 0; k0 < INPUT_DIM; k0 += 32) {
    bf16x8 a0 = *(const bf16x8*)(ah + k0);
    bf16x8 a1 = *(const bf16x8*)(ah + k0 + 8);
    bf16x16 a_hi = __builtin_shufflevector(a0, a1, 0, 1, 2, 3, 4, 5, 6, 7, 8,
                                           9, 10, 11, 12, 13, 14, 15);
    bf16x8 a2 = *(const bf16x8*)(al + k0);
    bf16x8 a3 = *(const bf16x8*)(al + k0 + 8);
    bf16x16 a_lo = __builtin_shufflevector(a2, a3, 0, 1, 2, 3, 4, 5, 6, 7, 8,
                                           9, 10, 11, 12, 13, 14, 15);
    bf16x16 b_hi = *(const bf16x16*)(bh + k0);
    bf16x16 b_lo = *(const bf16x16*)(bl + k0);

    acc = __builtin_amdgcn_wmma_f32_16x16x32_bf16(false, a_hi, false, b_hi,
                                                  (short)0, acc, false, false);
    acc = __builtin_amdgcn_wmma_f32_16x16x32_bf16(false, a_hi, false, b_lo,
                                                  (short)0, acc, false, false);
    acc = __builtin_amdgcn_wmma_f32_16x16x32_bf16(false, a_lo, false, b_hi,
                                                  (short)0, acc, false, false);
  }

#pragma unroll
  for (int j = 0; j < 8; ++j) {
    int row = tileM * 16 + half * 8 + j;
    xw[(size_t)row * NUM_HIDDEN + tileN * 16 + mn] = acc[j];
  }
}

// ---------------------------------------------------------------------------
// GEMM fallback (fused conversion, small-workspace path). Same math.
// ---------------------------------------------------------------------------
__global__ __launch_bounds__(256) void gemm_xw_fused_kernel(
    const float* __restrict__ x, const float* __restrict__ Whx,
    float* __restrict__ xw) {
  const int lane  = threadIdx.x & 31;
  const int wid   = (blockIdx.x * 256 + threadIdx.x) >> 5;
  const int tileM = wid >> 7;
  const int tileN = wid & 127;
  const int half  = lane >> 4;
  const int mn    = lane & 15;

  const int arow = tileM * 16 + mn;
  const int bcol = tileN * 16 + mn;

  floatx8 acc = {};
  for (int k0 = 0; k0 < INPUT_DIM; k0 += 32) {
    bf16x16 a_hi, a_lo, b_hi, b_lo;
    const float* aptr = x + (size_t)arow * INPUT_DIM + k0 + half * 16;
#pragma unroll
    for (int i = 0; i < 8; ++i) {
      float f0 = aptr[i];
      __bf16 h0 = f2bf_rne(f0);
      a_hi[i] = h0;
      a_lo[i] = f2bf_rne(f0 - bf2f(h0));
      float f1 = aptr[8 + i];
      __bf16 h1 = f2bf_rne(f1);
      a_hi[8 + i] = h1;
      a_lo[8 + i] = f2bf_rne(f1 - bf2f(h1));
    }
    const float* bptr = Whx + (size_t)(k0 + half * 16) * NUM_HIDDEN + bcol;
#pragma unroll
    for (int i = 0; i < 16; ++i) {
      float f = bptr[(size_t)i * NUM_HIDDEN];
      __bf16 h = f2bf_rne(f);
      b_hi[i] = h;
      b_lo[i] = f2bf_rne(f - bf2f(h));
    }
    acc = __builtin_amdgcn_wmma_f32_16x16x32_bf16(false, a_hi, false, b_hi,
                                                  (short)0, acc, false, false);
    acc = __builtin_amdgcn_wmma_f32_16x16x32_bf16(false, a_hi, false, b_lo,
                                                  (short)0, acc, false, false);
    acc = __builtin_amdgcn_wmma_f32_16x16x32_bf16(false, a_lo, false, b_hi,
                                                  (short)0, acc, false, false);
  }
#pragma unroll
  for (int j = 0; j < 8; ++j) {
    int row = tileM * 16 + half * 8 + j;
    xw[(size_t)row * NUM_HIDDEN + tileN * 16 + mn] = acc[j];
  }
}

// ---------------------------------------------------------------------------
// Persistent recurrence. 128 blocks x 256 threads; block b owns columns
// [16b, 16b+16), thread = ksub(0..15) x col_local(0..15): W_hh reads are
// coalesced, each thread does 128 FMAs/step. v staged in LDS; one monotonic
// atomic grid barrier per step (128 blocks trivially co-resident).
// ---------------------------------------------------------------------------
__global__ __launch_bounds__(256) void rnn_kernel(
    const float* __restrict__ xw, const float* __restrict__ Whh,
    const float* __restrict__ bh, float* __restrict__ v0,
    float* __restrict__ v1, unsigned* __restrict__ bar) {
  __shared__ float vin_s[NUM_HIDDEN];
  __shared__ float red[256];

  const int tid  = threadIdx.x;
  const int ksub = tid >> 4;                     // 0..15, 128 k each
  const int coll = tid & 15;                     // 0..15
  const int col  = blockIdx.x * 16 + coll;

  float* bufs[2] = {v0, v1};

  for (int t = 0; t < SEQ_LEN; ++t) {
    const float* vin = bufs[t & 1];
    float* vout      = bufs[(t + 1) & 1];

    for (int j = tid; j < NUM_HIDDEN; j += 256) vin_s[j] = vin[j];
    __syncthreads();

    float acc = 0.0f;
    const int k0 = ksub * 128;
    const float* wp = Whh + (size_t)k0 * NUM_HIDDEN + col;
#pragma unroll 8
    for (int k = 0; k < 128; ++k) {
      acc = fmaf(vin_s[k0 + k], wp[(size_t)k * NUM_HIDDEN], acc);
    }
    red[tid] = acc;
    __syncthreads();

    if (tid < 16) {                              // tid == coll here
      float s = 0.0f;
#pragma unroll
      for (int j = 0; j < 16; ++j) s += red[j * 16 + tid];
      vout[col] = tanhf(xw[(size_t)t * NUM_HIDDEN + col] + s + bh[col]);
    }

    // ---- grid barrier (monotonic counter, release/acquire, agent scope) ----
    __threadfence();
    __syncthreads();
    if (tid == 0) {
      __hip_atomic_fetch_add(bar, 1u, __ATOMIC_RELEASE,
                             __HIP_MEMORY_SCOPE_AGENT);
      const unsigned target = (unsigned)(t + 1) * RNN_BLOCKS;
      while (__hip_atomic_load(bar, __ATOMIC_ACQUIRE,
                               __HIP_MEMORY_SCOPE_AGENT) < target) {
        __builtin_amdgcn_s_sleep(2);
      }
    }
    __syncthreads();
    __threadfence();
  }
}

// ---------------------------------------------------------------------------
// logits[c] = 2048 * (v @ W_ph + b_p)[c]; softmax -> out[10].
// ---------------------------------------------------------------------------
__global__ __launch_bounds__(256) void out_kernel(
    const float* __restrict__ v, const float* __restrict__ Wph,
    const float* __restrict__ bp, float* __restrict__ out) {
  __shared__ float red[NUM_CLASSES * 256];
  const int tid = threadIdx.x;

  float acc[NUM_CLASSES];
#pragma unroll
  for (int c = 0; c < NUM_CLASSES; ++c) acc[c] = 0.0f;

  for (int k = tid; k < NUM_HIDDEN; k += 256) {
    float vk = v[k];
#pragma unroll
    for (int c = 0; c < NUM_CLASSES; ++c)
      acc[c] = fmaf(vk, Wph[k * NUM_CLASSES + c], acc[c]);
  }
#pragma unroll
  for (int c = 0; c < NUM_CLASSES; ++c) red[c * 256 + tid] = acc[c];
  __syncthreads();

  for (int s = 128; s > 0; s >>= 1) {
    if (tid < s) {
#pragma unroll
      for (int c = 0; c < NUM_CLASSES; ++c)
        red[c * 256 + tid] += red[c * 256 + tid + s];
    }
    __syncthreads();
  }

  if (tid == 0) {
    float logits[NUM_CLASSES];
    float m = -__builtin_inff();
    for (int c = 0; c < NUM_CLASSES; ++c) {
      logits[c] = (float)NUM_HIDDEN * (red[c * 256] + bp[c]);
      m = fmaxf(m, logits[c]);
    }
    float s = 0.0f;
    for (int c = 0; c < NUM_CLASSES; ++c) {
      logits[c] = expf(logits[c] - m);
      s += logits[c];
    }
    for (int c = 0; c < NUM_CLASSES; ++c) out[c] = logits[c] / s;
  }
}

// ---------------------------------------------------------------------------
// Host launcher. Inputs: x, W_hx, W_hh, W_ph, b_h, b_p (all float32).
// Workspace: [xw 2MB][v0|v1|bar][@4MB x_hi 2MB][@6MB x_lo 2MB]
//            [@8MB WT_hi 16MB][@24MB WT_lo 16MB]  (split path needs 40MB)
// ---------------------------------------------------------------------------
extern "C" void kernel_launch(void* const* d_in, const int* in_sizes, int n_in,
                              void* d_out, int out_size, void* d_ws,
                              size_t ws_size, hipStream_t stream) {
  const float* x   = (const float*)d_in[0];
  const float* Whx = (const float*)d_in[1];
  const float* Whh = (const float*)d_in[2];
  const float* Wph = (const float*)d_in[3];
  const float* bh  = (const float*)d_in[4];
  const float* bp  = (const float*)d_in[5];
  float* out = (float*)d_out;

  char* ws = (char*)d_ws;
  const size_t MB = 1024u * 1024u;
  float* xw     = (float*)ws;
  float* v0     = (float*)(ws + 2 * MB);
  float* v1     = v0 + NUM_HIDDEN;
  unsigned* bar = (unsigned*)(v1 + NUM_HIDDEN);
  __bf16* xhi   = (__bf16*)(ws + 4 * MB);
  __bf16* xlo   = (__bf16*)(ws + 6 * MB);
  __bf16* wThi  = (__bf16*)(ws + 8 * MB);
  __bf16* wTlo  = (__bf16*)(ws + 24 * MB);

  init_kernel<<<8, 256, 0, stream>>>(v0, bar);

  if (ws_size >= 40 * MB) {
    split_x_kernel<<<(SEQ_LEN * INPUT_DIM) / 256, 256, 0, stream>>>(x, xhi,
                                                                    xlo);
    split_wT_kernel<<<(INPUT_DIM / 32) * (NUM_HIDDEN / 32), 256, 0, stream>>>(
        Whx, wThi, wTlo);
    gemm_xw_bf16_kernel<<<256, 256, 0, stream>>>(xhi, xlo, wThi, wTlo, xw);
  } else {
    gemm_xw_fused_kernel<<<256, 256, 0, stream>>>(x, Whx, xw);
  }

  rnn_kernel<<<RNN_BLOCKS, 256, 0, stream>>>(xw, Whh, bh, v0, v1, bar);
  // final v is in bufs[(255+1)&1] == v0
  out_kernel<<<1, 256, 0, stream>>>(v0, Wph, bp, out);
}
